// multihead_self_attention_36275293782759
// MI455X (gfx1250) — matrix operand order repaired and checked by reading, not compiled, not run on hardware
//
#include <hip/hip_runtime.h>
#include <hip/hip_bf16.h>

typedef __bf16 bf16_t;
typedef __attribute__((ext_vector_type(16))) __bf16 v16bf;
typedef __attribute__((ext_vector_type(8)))  __bf16 v8bf;   // 16 bytes
typedef __attribute__((ext_vector_type(8)))  float  v8f;

#define D_MODEL  1024
#define NUM_HEADS 16
#define D_H       64
#define BATCH      4
#define SEQ     2048
#define M_ROWS  (BATCH * SEQ)

#define P_STRIDE 40   // padded LDS row stride (elements) for the P tile
#define PF_DIST  512  // prefetch distance in elements (1 KB)

union AFrag { v16bf v; v8bf h[2]; };

static __device__ __forceinline__ v8f wmma_bf16(v16bf a, v16bf b, v8f c) {
  // D = A(16x32 bf16) * B(32x16 bf16) + C(16x16 f32)
  return __builtin_amdgcn_wmma_f32_16x16x32_bf16(false, a, false, b, (short)0, c,
                                                 false, false);
}

static __device__ __forceinline__ v8f zero_v8f() {
  v8f z;
  #pragma unroll
  for (int i = 0; i < 8; ++i) z[i] = 0.0f;
  return z;
}

// Wave-local LDS fence: DS pipe is in-order per wave; this just stops the
// compiler from reordering around the P-tile store->reload and drains DScnt.
static __device__ __forceinline__ void lds_fence_wave() {
  asm volatile("s_wait_dscnt 0x0" ::: "memory");
}

// ---------------------------------------------------------------------------
// fp32 -> bf16 conversion
// ---------------------------------------------------------------------------
__global__ void cvt_f32_bf16(const float* __restrict__ src,
                             bf16_t* __restrict__ dst, int n) {
  int i = blockIdx.x * blockDim.x + threadIdx.x;
  if (i < n) dst[i] = (bf16_t)src[i];
}

// ---------------------------------------------------------------------------
// Shared 32x64 wave-tile GEMM k-step: 2 A-frags x 4 B-frags -> 8 WMMAs.
// C[m, n] = sum_k Arow[m, k] * Brow[n, k]   (both row-major along k)
// ---------------------------------------------------------------------------
static __device__ __forceinline__ void gemm_kstep(
    int k0, bool pf, int half, int l15, int n0,
    const bf16_t* __restrict__ arow0, const bf16_t* __restrict__ arow1,
    const bf16_t* __restrict__ W, v8f acc[2][4]) {
  if (pf) {
    __builtin_prefetch(arow0 + k0 + PF_DIST, 0, 1);
    __builtin_prefetch(arow1 + k0 + PF_DIST, 0, 1);
  }
  AFrag a0, a1;
  a0.h[0] = *(const v8bf*)(arow0 + k0 + half * 8);
  a0.h[1] = *(const v8bf*)(arow0 + k0 + 16 + half * 8);
  a1.h[0] = *(const v8bf*)(arow1 + k0 + half * 8);
  a1.h[1] = *(const v8bf*)(arow1 + k0 + 16 + half * 8);
  #pragma unroll
  for (int j = 0; j < 4; ++j) {
    const bf16_t* wrow = W + (size_t)(n0 + j * 16 + l15) * D_MODEL
                           + k0 + half * 8;
    AFrag bfr;
    bfr.h[0] = *(const v8bf*)(wrow);
    bfr.h[1] = *(const v8bf*)(wrow + 16);
    acc[0][j] = wmma_bf16(a0.v, bfr.v, acc[0][j]);
    acc[1][j] = wmma_bf16(a1.v, bfr.v, acc[1][j]);
  }
}

// ---------------------------------------------------------------------------
// QKV projection: out[m, o] = sum_i X[m, i] * W[o, i]   (C = X * W^T)
// z = blockIdx.z selects {Wq, Wk, Wv}.  Q,K stored [B,H,S,Dh]; V stored
// transposed [B,H,Dh,S] so the attention P*V B-fragments are contiguous.
// ---------------------------------------------------------------------------
__global__ __launch_bounds__(128)
void qkv_gemm(const bf16_t* __restrict__ X,
              const bf16_t* __restrict__ Wq, const bf16_t* __restrict__ Wk,
              const bf16_t* __restrict__ Wv,
              bf16_t* __restrict__ Q, bf16_t* __restrict__ K,
              bf16_t* __restrict__ Vt) {
  const int lane = threadIdx.x & 31;
  const int wave = threadIdx.x >> 5;
  const int half = lane >> 4;
  const int l15  = lane & 15;
  const int m0   = (blockIdx.x * 4 + wave) * 32;
  const int n0   = blockIdx.y * 64;
  const int z    = blockIdx.z;
  const bf16_t* W = (z == 0) ? Wq : (z == 1) ? Wk : Wv;

  const bf16_t* xrow0 = X + (size_t)(m0 + l15) * D_MODEL;
  const bf16_t* xrow1 = xrow0 + (size_t)16 * D_MODEL;

  v8f acc[2][4];
  #pragma unroll
  for (int i = 0; i < 2; ++i)
    #pragma unroll
    for (int j = 0; j < 4; ++j) acc[i][j] = zero_v8f();

  // split loop: prefetching half, then plain half (no in-loop guard)
  for (int k0 = 0; k0 < D_MODEL - PF_DIST; k0 += 32)
    gemm_kstep(k0, true, half, l15, n0, xrow0, xrow1, W, acc);
  for (int k0 = D_MODEL - PF_DIST; k0 < D_MODEL; k0 += 32)
    gemm_kstep(k0, false, half, l15, n0, xrow0, xrow1, W, acc);

  const int head = blockIdx.y;            // n-tile width 64 == D_H
  const int b    = m0 / SEQ;
  const int s0   = m0 - b * SEQ;
  const size_t bh = (size_t)b * NUM_HEADS + head;

  if (z < 2) {                            // Q or K: [B,H,S,Dh]
    bf16_t* dst = (z == 0 ? Q : K) + bh * SEQ * D_H;
    #pragma unroll
    for (int i = 0; i < 2; ++i)
      #pragma unroll
      for (int j = 0; j < 4; ++j) {
        int dcol = j * 16 + l15;
        #pragma unroll
        for (int r = 0; r < 8; ++r) {
          int s = s0 + i * 16 + r + 8 * half;
          dst[(size_t)s * D_H + dcol] = (bf16_t)acc[i][j][r];
        }
      }
  } else {                                // V transposed: [B,H,Dh,S]
    bf16_t* dst = Vt + bh * D_H * SEQ;
    #pragma unroll
    for (int i = 0; i < 2; ++i)
      #pragma unroll
      for (int j = 0; j < 4; ++j) {
        int dcol = j * 16 + l15;
        v8bf pk;
        #pragma unroll
        for (int r = 0; r < 8; ++r) pk[r] = (bf16_t)acc[i][j][r];
        *(v8bf*)(dst + (size_t)dcol * SEQ + s0 + i * 16 + 8 * half) = pk;
      }
  }
}

// ---------------------------------------------------------------------------
// Causal flash attention: 4 independent waves per block, one 16-query tile
// each.  Online softmax; per-row stats live in per-lane scalar arrays matching
// the WMMA C-layout (row r+8*half in element r).  P re-layout C->A goes via a
// wave-private padded LDS tile (no cross-wave barrier: trip counts diverge).
// ---------------------------------------------------------------------------
__global__ __launch_bounds__(128)
void flash_attn(const bf16_t* __restrict__ Q, const bf16_t* __restrict__ K,
                const bf16_t* __restrict__ Vt, bf16_t* __restrict__ O) {
  __shared__ __align__(16) bf16_t PlAll[4][16 * P_STRIDE];

  const int lane = threadIdx.x & 31;
  const int wave = threadIdx.x >> 5;
  const int half = lane >> 4;
  const int l15  = lane & 15;
  const int q0   = blockIdx.x * 64 + wave * 16;
  const int bh   = blockIdx.y;
  const int b    = bh / NUM_HEADS;
  const int h    = bh % NUM_HEADS;
  bf16_t* Pl = PlAll[wave];

  const bf16_t* Qp  = Q  + (size_t)bh * SEQ * D_H;
  const bf16_t* Kp  = K  + (size_t)bh * SEQ * D_H;
  const bf16_t* Vtp = Vt + (size_t)bh * D_H * SEQ;

  // Q tile as two A-fragments (d = 0..31, 32..63)
  AFrag aq[2];
  const bf16_t* qrow = Qp + (size_t)(q0 + l15) * D_H;
  #pragma unroll
  for (int t = 0; t < 2; ++t) {
    aq[t].h[0] = *(const v8bf*)(qrow + t * 32 + half * 8);
    aq[t].h[1] = *(const v8bf*)(qrow + t * 32 + 16 + half * 8);
  }

  float rowmax[8], rowsum[8];
  #pragma unroll
  for (int r = 0; r < 8; ++r) { rowmax[r] = -1e30f; rowsum[r] = 0.0f; }
  v8f oacc[4];
  #pragma unroll
  for (int j = 0; j < 4; ++j) oacc[j] = zero_v8f();

  const float scale = 0.125f;  // 1/sqrt(64)

  for (int n0 = 0; n0 < q0 + 16; n0 += 32) {
    // S = Q * K^T : two 16x16 score tiles (cols n0..+15, n0+16..+31)
    v8f sc[2];
    sc[0] = zero_v8f(); sc[1] = zero_v8f();
    #pragma unroll
    for (int c = 0; c < 2; ++c) {
      #pragma unroll
      for (int t = 0; t < 2; ++t) {
        const bf16_t* krow = Kp + (size_t)(n0 + c * 16 + l15) * D_H
                               + t * 32 + half * 8;
        AFrag bk;
        bk.h[0] = *(const v8bf*)(krow);
        bk.h[1] = *(const v8bf*)(krow + 16);
        sc[c] = wmma_bf16(aq[t].v, bk.v, sc[c]);
      }
    }

    // causal mask + online softmax (row r+8*half lives in element r)
    float corr[8];
    #pragma unroll
    for (int r = 0; r < 8; ++r) {
      int mg  = q0 + r + 8 * half;
      int ng0 = n0 + l15;
      int ng1 = n0 + 16 + l15;
      float v0 = (ng0 <= mg) ? sc[0][r] * scale : -1e30f;
      float v1 = (ng1 <= mg) ? sc[1][r] * scale : -1e30f;
      float tm = fmaxf(v0, v1);
      #pragma unroll
      for (int off = 8; off >= 1; off >>= 1)
        tm = fmaxf(tm, __shfl_xor(tm, off, 32));   // stays within 16-lane half
      float nm = fmaxf(rowmax[r], tm);
      corr[r]  = __expf(rowmax[r] - nm);
      rowmax[r] = nm;
      float p0 = __expf(v0 - nm);
      float p1 = __expf(v1 - nm);
      float ps = p0 + p1;
      #pragma unroll
      for (int off = 8; off >= 1; off >>= 1)
        ps += __shfl_xor(ps, off, 32);
      rowsum[r] = rowsum[r] * corr[r] + ps;
      int m = r + 8 * half;
      Pl[m * P_STRIDE + l15]      = (bf16_t)p0;
      Pl[m * P_STRIDE + 16 + l15] = (bf16_t)p1;
    }

    // rescale running output by exp(old_max - new_max)
    #pragma unroll
    for (int j = 0; j < 4; ++j)
      #pragma unroll
      for (int r = 0; r < 8; ++r) oacc[j][r] *= corr[r];

    lds_fence_wave();                     // wave-private tile, DS in-order
    AFrag ap;
    ap.h[0] = *(const v8bf*)(&Pl[l15 * P_STRIDE + half * 8]);
    ap.h[1] = *(const v8bf*)(&Pl[l15 * P_STRIDE + 16 + half * 8]);

    // O += P * V  (Vt rows contiguous along s)
    #pragma unroll
    for (int j = 0; j < 4; ++j) {
      const bf16_t* vrow = Vtp + (size_t)(j * 16 + l15) * SEQ + n0 + half * 8;
      AFrag bv;
      bv.h[0] = *(const v8bf*)(vrow);
      bv.h[1] = *(const v8bf*)(vrow + 16);
      oacc[j] = wmma_bf16(ap.v, bv.v, oacc[j]);
    }
  }

  // normalize and store merged-head bf16 output [B*S, D]
  float inv[8];
  #pragma unroll
  for (int r = 0; r < 8; ++r) inv[r] = 1.0f / rowsum[r];
  #pragma unroll
  for (int j = 0; j < 4; ++j) {
    int col = h * D_H + j * 16 + l15;
    #pragma unroll
    for (int r = 0; r < 8; ++r) {
      int s = q0 + r + 8 * half;
      O[((size_t)b * SEQ + s) * D_MODEL + col] = (bf16_t)(oacc[j][r] * inv[r]);
    }
  }
}

// ---------------------------------------------------------------------------
// Output projection: out[m, o] = sum_i A[m, i] * Wo[o, i]  -> fp32 d_out
// ---------------------------------------------------------------------------
__global__ __launch_bounds__(128)
void out_gemm(const bf16_t* __restrict__ A, const bf16_t* __restrict__ Wo,
              float* __restrict__ out) {
  const int lane = threadIdx.x & 31;
  const int wave = threadIdx.x >> 5;
  const int half = lane >> 4;
  const int l15  = lane & 15;
  const int m0   = (blockIdx.x * 4 + wave) * 32;
  const int n0   = blockIdx.y * 64;

  const bf16_t* arow0 = A + (size_t)(m0 + l15) * D_MODEL;
  const bf16_t* arow1 = arow0 + (size_t)16 * D_MODEL;

  v8f acc[2][4];
  #pragma unroll
  for (int i = 0; i < 2; ++i)
    #pragma unroll
    for (int j = 0; j < 4; ++j) acc[i][j] = zero_v8f();

  for (int k0 = 0; k0 < D_MODEL - PF_DIST; k0 += 32)
    gemm_kstep(k0, true, half, l15, n0, arow0, arow1, Wo, acc);
  for (int k0 = D_MODEL - PF_DIST; k0 < D_MODEL; k0 += 32)
    gemm_kstep(k0, false, half, l15, n0, arow0, arow1, Wo, acc);

  #pragma unroll
  for (int i = 0; i < 2; ++i)
    #pragma unroll
    for (int j = 0; j < 4; ++j) {
      int col = n0 + j * 16 + l15;
      #pragma unroll
      for (int r = 0; r < 8; ++r) {
        int m = m0 + i * 16 + r + 8 * half;
        out[(size_t)m * D_MODEL + col] = acc[i][j][r];
      }
    }
}

// ---------------------------------------------------------------------------
extern "C" void kernel_launch(void* const* d_in, const int* in_sizes, int n_in,
                              void* d_out, int out_size, void* d_ws, size_t ws_size,
                              hipStream_t stream) {
  (void)in_sizes; (void)n_in; (void)out_size; (void)ws_size;

  const float* x  = (const float*)d_in[0];
  const float* Wq = (const float*)d_in[1];
  const float* Wk = (const float*)d_in[2];
  const float* Wv = (const float*)d_in[3];
  const float* Wo = (const float*)d_in[4];

  char* ws = (char*)d_ws;
  size_t off = 0;
  bf16_t* xb  = (bf16_t*)(ws + off); off += (size_t)M_ROWS * D_MODEL * sizeof(bf16_t);
  bf16_t* wqb = (bf16_t*)(ws + off); off += (size_t)D_MODEL * D_MODEL * sizeof(bf16_t);
  bf16_t* wkb = (bf16_t*)(ws + off); off += (size_t)D_MODEL * D_MODEL * sizeof(bf16_t);
  bf16_t* wvb = (bf16_t*)(ws + off); off += (size_t)D_MODEL * D_MODEL * sizeof(bf16_t);
  bf16_t* wob = (bf16_t*)(ws + off); off += (size_t)D_MODEL * D_MODEL * sizeof(bf16_t);
  bf16_t* Qb  = (bf16_t*)(ws + off); off += (size_t)M_ROWS * D_MODEL * sizeof(bf16_t);
  bf16_t* Kb  = (bf16_t*)(ws + off); off += (size_t)M_ROWS * D_MODEL * sizeof(bf16_t);
  bf16_t* Vtb = (bf16_t*)(ws + off); off += (size_t)M_ROWS * D_MODEL * sizeof(bf16_t);
  bf16_t* AOb = (bf16_t*)(ws + off); off += (size_t)M_ROWS * D_MODEL * sizeof(bf16_t);

  const int nx = M_ROWS * D_MODEL;
  const int nw = D_MODEL * D_MODEL;
  cvt_f32_bf16<<<(nx + 255) / 256, 256, 0, stream>>>(x,  xb,  nx);
  cvt_f32_bf16<<<(nw + 255) / 256, 256, 0, stream>>>(Wq, wqb, nw);
  cvt_f32_bf16<<<(nw + 255) / 256, 256, 0, stream>>>(Wk, wkb, nw);
  cvt_f32_bf16<<<(nw + 255) / 256, 256, 0, stream>>>(Wv, wvb, nw);
  cvt_f32_bf16<<<(nw + 255) / 256, 256, 0, stream>>>(Wo, wob, nw);

  qkv_gemm<<<dim3(M_ROWS / 128, D_MODEL / 64, 3), 128, 0, stream>>>(
      xb, wqb, wkb, wvb, Qb, Kb, Vtb);

  flash_attn<<<dim3(SEQ / 64, BATCH * NUM_HEADS), 128, 0, stream>>>(
      Qb, Kb, Vtb, AOb);

  out_gemm<<<dim3(M_ROWS / 128, D_MODEL / 64), 128, 0, stream>>>(
      AOb, wob, (float*)d_out);
}
